// Encoder_14594298872396
// MI455X (gfx1250) — matrix operand order repaired and checked by reading, not compiled, not run on hardware
//
#include <hip/hip_runtime.h>

// Problem dims (fixed by the reference)
#define B_  256
#define T_  128
#define N_  1024
#define H_  1024
#define G4  4096   // 4*H
#define K2  2048   // N + H (concat GEMM K)

typedef __attribute__((ext_vector_type(16))) _Float16 v16h;
typedef __attribute__((ext_vector_type(8)))  float    v8f;

// ---------------------------------------------------------------------------
// score_x[b,n] = sum_t x[b,t,n] * w_x[t] + attn_b   (w_x = attn_w[2H:])
__global__ void k_scorex(const float* __restrict__ x,
                         const float* __restrict__ attn_w,
                         const float* __restrict__ attn_b,
                         float* __restrict__ score) {
    int idx = blockIdx.x * blockDim.x + threadIdx.x;   // B*N threads
    int b = idx >> 10, n = idx & 1023;
    const float* xp = x + (size_t)b * T_ * N_ + n;
    float s = attn_b[0];
#pragma unroll 4
    for (int t = 0; t < T_; ++t)
        s += xp[(size_t)t * N_] * attn_w[2 * H_ + t];
    score[idx] = s;
}

// ---------------------------------------------------------------------------
// In-place softmax over each row of length N (a = softmax(score_x, axis=1)).
// (Attention weights are time-invariant: softmax is shift-invariant w.r.t.
//  the per-row h/c recurrent constant, so it is computed once.)
__global__ void k_softmax(float* __restrict__ score) {
    __shared__ float red[256];
    int b = blockIdx.x, tid = threadIdx.x;
    float* row = score + (size_t)b * N_;
    float m = -1e30f;
    for (int n = tid; n < N_; n += 256) m = fmaxf(m, row[n]);
    red[tid] = m; __syncthreads();
    for (int s = 128; s > 0; s >>= 1) {
        if (tid < s) red[tid] = fmaxf(red[tid], red[tid + s]);
        __syncthreads();
    }
    m = red[0]; __syncthreads();
    float sum = 0.f;
    for (int n = tid; n < N_; n += 256) sum += expf(row[n] - m);
    red[tid] = sum; __syncthreads();
    for (int s = 128; s > 0; s >>= 1) {
        if (tid < s) red[tid] += red[tid + s];
        __syncthreads();
    }
    float inv = 1.0f / red[0];
    for (int n = tid; n < N_; n += 256) row[n] = expf(row[n] - m) * inv;
}

// ---------------------------------------------------------------------------
// input_weighted[b,t,n] = a[b,n] * x[b,t,n]
__global__ void k_wi(const float* __restrict__ x, const float* __restrict__ a,
                     float* __restrict__ iw) {
    size_t idx = (size_t)blockIdx.x * blockDim.x + threadIdx.x;  // B*T*N
    int n = (int)(idx & 1023);
    int b = (int)(idx >> 17);            // T*N = 2^17
    iw[idx] = a[(size_t)b * N_ + n] * x[idx];
}

// ---------------------------------------------------------------------------
// bias_sum[j] = b_ih[j] + b_hh[j]  (4096 wide, gate-major)
__global__ void k_bias(const float* __restrict__ bih, const float* __restrict__ bhh,
                       float* __restrict__ bias) {
    int j = blockIdx.x * blockDim.x + threadIdx.x;
    bias[j] = bih[j] + bhh[j];
}

// ---------------------------------------------------------------------------
// Pack Wcat = [W_ih | W_hh] (4096 x 2048) into WMMA B-fragment order:
//   wpk[((kc*256 + ct)*32 + lane)*16 + p] = Wcat[ct*16 + lane%16]
//                                               [kc*32 + 16*(lane/16) + p]
__global__ void k_pack(const float* __restrict__ Wih, const float* __restrict__ Whh,
                       _Float16* __restrict__ wpk) {
    long i = (long)blockIdx.x * blockDim.x + threadIdx.x;  // 2048*4096 halfs
    int p    = (int)(i & 15);
    int lane = (int)((i >> 4) & 31);
    int ct   = (int)((i >> 9) & 255);
    int kc   = (int)(i >> 17);
    int k = kc * 32 + ((lane >> 4) << 4) + p;
    int j = ct * 16 + (lane & 15);
    float w = (k < N_) ? Wih[(size_t)j * N_ + k]
                       : Whh[(size_t)j * H_ + (k - N_)];
    wpk[i] = (_Float16)w;
}

// ---------------------------------------------------------------------------
// Zero c (fp32) and h0 (f16).
__global__ void k_init(float* __restrict__ c, _Float16* __restrict__ h0) {
    int idx = blockIdx.x * blockDim.x + threadIdx.x;   // B*H
    c[idx] = 0.f;
    h0[idx] = (_Float16)0.f;
}

// ---------------------------------------------------------------------------
// Fused LSTM step: gates GEMM (f16 WMMA, K=2048) + bias + pointwise update.
//
// Grid (8 row-groups, 16 col-groups), block 128 = 4 waves. Each wave owns
// column tile ctb = blockIdx.y*4 + wave (j = ctb*16 + lane%16) and computes
// the FOUR gate tiles {i,f,g,o} at columns {j, H+j, 2H+j, 3H+j} for TWO
// 16-row tiles (M=32): one B fragment feeds two WMMAs, halving B traffic.
// The LSTM pointwise update is local to the wave's accumulators.
//
// The 32 x 2048 f16 A tile (x-half converted fp32->f16 from input_weighted,
// h-half from the previous step's f16 h buffer) is staged once in LDS
// (128 KB), XOR-swizzled on 16B chunks for conflict-free ds_load_b128 reads.
__global__ void __launch_bounds__(128)
k_step(const float* __restrict__ iw,        // (B,T,N) input_weighted
       const _Float16* __restrict__ hin,    // (B,H) f16 h from prev step
       const _Float16* __restrict__ wpk,    // packed weights
       const float* __restrict__ bias,      // 4096
       float* __restrict__ cbuf,            // (B,H) fp32 cell state
       _Float16* __restrict__ hout,         // (B,H) f16 h for next step
       float* __restrict__ out_ie,          // (B,T,H) input_encoded
       int t) {
    __shared__ _Float16 lds[32 * 2048];     // 128 KB A tile (swizzled)

    int tid  = threadIdx.x;
    int lane = tid & 31;
    int wave = tid >> 5;
    int rg   = blockIdx.x;                  // row group: batch rows rg*32..+31
    int ctb  = blockIdx.y * 4 + wave;       // 0..63 col tile within a gate
    int lh   = lane >> 4;
    int ll   = lane & 15;

    // ---- Stage A tile into LDS -------------------------------------------
    // x half: 32 rows x 1024 fp32 -> f16. 8192 float4 chunks, 64 iters.
    {
        union { _Float16 h[4]; uint2 u; } pk;
        for (int it = 0; it < 64; ++it) {
            int flat = it * 128 + tid;
            int row  = flat >> 8;            // 256 float4 per row
            int q4   = flat & 255;
            const float4 v = *(const float4*)(iw +
                ((size_t)(rg * 32 + row) * T_ + t) * N_ + q4 * 4);
            pk.h[0] = (_Float16)v.x; pk.h[1] = (_Float16)v.y;
            pk.h[2] = (_Float16)v.z; pk.h[3] = (_Float16)v.w;
            int c8 = q4 >> 1;                // 16B chunk index within row
            int idx = row * 2048 + ((c8 ^ (row & 15)) << 3) + ((q4 & 1) << 2);
            *(uint2*)(lds + idx) = pk.u;
        }
        // h half: 32 rows x 1024 f16. 4096 uint4 chunks, 32 iters.
        for (int it = 0; it < 32; ++it) {
            int flat = it * 128 + tid;
            int row  = flat >> 7;            // 128 uint4 per row
            int q    = flat & 127;
            uint4 v = *(const uint4*)(hin + (size_t)(rg * 32 + row) * H_ + q * 8);
            int c8 = 128 + q;
            int idx = row * 2048 + ((c8 ^ (row & 15)) << 3);
            *(uint4*)(lds + idx) = v;
        }
    }
    __syncthreads();

    // ---- Main K loop: 2 A fragments -> 8 WMMAs (4 gates x 2 row tiles) ---
    const _Float16* b0 = wpk + ((size_t)(0 * 64 + ctb) * 32 + lane) * 16;
    const _Float16* b1 = wpk + ((size_t)(1 * 64 + ctb) * 32 + lane) * 16;
    const _Float16* b2 = wpk + ((size_t)(2 * 64 + ctb) * 32 + lane) * 16;
    const _Float16* b3 = wpk + ((size_t)(3 * 64 + ctb) * 32 + lane) * 16;

    v8f ci0 = {}, cf0 = {}, cg0 = {}, co0 = {};
    v8f ci1 = {}, cf1 = {}, cg1 = {}, co1 = {};
    union AB { v16h v; uint4 q[2]; };
    AB A0e, A1e, A0o, A1o, Bf;

#define LOAD_A(Adst, m, kcv)                                                 \
    {                                                                        \
        int cb = (kcv) * 4 + 2 * lh;                                         \
        int base = ((m) * 16 + ll) * 2048;                                   \
        (Adst).q[0] = *(const uint4*)(lds + base + ((cb ^ ll) << 3));        \
        (Adst).q[1] = *(const uint4*)(lds + base + (((cb + 1) ^ ll) << 3));  \
    }
#define GATE(bp, acc0, acc1, Aa, Ab, ko)                                     \
    {                                                                        \
        Bf.q[0] = *(const uint4*)((bp) + (ko));                              \
        Bf.q[1] = *(const uint4*)((bp) + (ko) + 8);                          \
        acc0 = __builtin_amdgcn_wmma_f32_16x16x32_f16(false, (Aa).v, false,  \
                   Bf.v, (short)0, acc0, false, false);                      \
        acc1 = __builtin_amdgcn_wmma_f32_16x16x32_f16(false, (Ab).v, false,  \
                   Bf.v, (short)0, acc1, false, false);                      \
    }

    for (int kc = 0; kc < 64; kc += 2) {
        size_t ko = (size_t)kc * (256 * 512);
        LOAD_A(A0e, 0, kc);
        LOAD_A(A1e, 1, kc);
        GATE(b0, ci0, ci1, A0e, A1e, ko);
        GATE(b1, cf0, cf1, A0e, A1e, ko);
        GATE(b2, cg0, cg1, A0e, A1e, ko);
        GATE(b3, co0, co1, A0e, A1e, ko);
        size_t ko2 = ko + (256 * 512);
        LOAD_A(A0o, 0, kc + 1);
        LOAD_A(A1o, 1, kc + 1);
        GATE(b0, ci0, ci1, A0o, A1o, ko2);
        GATE(b1, cf0, cf1, A0o, A1o, ko2);
        GATE(b2, cg0, cg1, A0o, A1o, ko2);
        GATE(b3, co0, co1, A0o, A1o, ko2);
    }
#undef LOAD_A
#undef GATE

    // ---- Epilogue: bias + LSTM pointwise, write c/h/out ------------------
    // D layout: VGPR r -> local row m*16 + r + 8*lh, col = ctb*16 + lane%16
    int j  = ctb * 16 + ll;
    float bi = bias[j], bf = bias[H_ + j], bg = bias[2 * H_ + j], bo = bias[3 * H_ + j];
#pragma unroll
    for (int m = 0; m < 2; ++m) {
        const v8f& ci = m ? ci1 : ci0;
        const v8f& cf = m ? cf1 : cf0;
        const v8f& cg = m ? cg1 : cg0;
        const v8f& co = m ? co1 : co0;
#pragma unroll
        for (int r = 0; r < 8; ++r) {
            int row = rg * 32 + m * 16 + lh * 8 + r;
            size_t cidx = (size_t)row * H_ + j;
            float gi = ci[r] + bi;
            float gf = cf[r] + bf;
            float gg = cg[r] + bg;
            float go = co[r] + bo;
            float si = 1.f / (1.f + expf(-gi));
            float sf = 1.f / (1.f + expf(-gf));
            float so = 1.f / (1.f + expf(-go));
            float cn = sf * cbuf[cidx] + si * tanhf(gg);
            float hn = so * tanhf(cn);
            cbuf[cidx] = cn;
            hout[cidx] = (_Float16)hn;
            out_ie[((size_t)row * T_ + t) * H_ + j] = hn;
        }
    }
}

// ---------------------------------------------------------------------------
extern "C" void kernel_launch(void* const* d_in, const int* in_sizes, int n_in,
                              void* d_out, int out_size, void* d_ws, size_t ws_size,
                              hipStream_t stream) {
    const float* x      = (const float*)d_in[0];  // (B,T,N)
    const float* W_ih   = (const float*)d_in[1];  // (4H,N)
    const float* W_hh   = (const float*)d_in[2];  // (4H,H)
    const float* b_ih   = (const float*)d_in[3];
    const float* b_hh   = (const float*)d_in[4];
    const float* attn_w = (const float*)d_in[5];
    const float* attn_b = (const float*)d_in[6];

    float* out_iw = (float*)d_out;                       // (B,T,N)
    float* out_ie = out_iw + (size_t)B_ * T_ * N_;       // (B,T,H)

    // Workspace carve-up (~19 MB)
    char* ws = (char*)d_ws;
    float*    score = (float*)ws;     ws += (size_t)B_ * N_ * 4;   // -> a
    float*    bias  = (float*)ws;     ws += (size_t)G4 * 4;
    float*    cbuf  = (float*)ws;     ws += (size_t)B_ * H_ * 4;
    _Float16* hA    = (_Float16*)ws;  ws += (size_t)B_ * H_ * 2;
    _Float16* hB    = (_Float16*)ws;  ws += (size_t)B_ * H_ * 2;
    _Float16* wpk   = (_Float16*)ws;  /* 2048*4096*2 = 16 MB */

    // One-time preprocessing
    k_scorex <<<(B_ * N_) / 256, 256, 0, stream>>>(x, attn_w, attn_b, score);
    k_softmax<<<B_, 256, 0, stream>>>(score);
    k_bias   <<<G4 / 256, 256, 0, stream>>>(b_ih, b_hh, bias);
    k_pack   <<<((size_t)K2 * G4) / 256, 256, 0, stream>>>(W_ih, W_hh, wpk);
    k_wi     <<<((size_t)B_ * T_ * N_) / 256, 256, 0, stream>>>(x, score, out_iw);
    k_init   <<<(B_ * H_) / 256, 256, 0, stream>>>(cbuf, hA);

    // Sequential LSTM over T: one fused kernel per step, h ping-pong.
    for (int t = 0; t < T_; ++t) {
        _Float16* hin  = (t & 1) ? hB : hA;
        _Float16* hout = (t & 1) ? hA : hB;
        k_step<<<dim3(8, 16), 128, 0, stream>>>(out_iw, hin, wpk, bias,
                                                cbuf, hout, out_ie, t);
    }
}
